// Conv_KNRM_80616536146801
// MI455X (gfx1250) — matrix operand order repaired and checked by reading, not compiled, not run on hardware
//
#include <hip/hip_runtime.h>
#include <hip/hip_bf16.h>
#include <stdint.h>

typedef __attribute__((ext_vector_type(16))) _Float16 v16h;
typedef __attribute__((ext_vector_type(8)))  float    v8f;
typedef __attribute__((ext_vector_type(2)))  _Float16 h2t;

union FragU { uint4 u[2]; v16h v; };

#define NB   32      // batch
#define NLQ  32      // query length
#define NLD  256     // doc length
#define NE   300     // embedding dim
#define NC   128     // conv channels
#define NK   11      // RBF kernels

// K padding per gram (k*300 rounded up to multiple of 32): 320 / 608 / 928

// ---- workspace layout (bytes) ----
static constexpr size_t OFF_FEATS = 0;                       // 32*99 f32
static constexpr size_t OFF_W     = 12800;                   // 1856*128 f16
static constexpr size_t OFF_XD    = OFF_W  + 475136;         // 30,408,704 B
static constexpr size_t OFF_XQ    = OFF_XD + 30408704;       //  3,801,088 B
static constexpr size_t OFF_DG    = OFF_XQ + 3801088;        //  6,291,456 B
static constexpr size_t OFF_QG    = OFF_DG + 6291456;        //    786,432 B

__device__ __constant__ float d_MU[NK]  = {1.0f,0.9f,0.7f,0.5f,0.3f,0.1f,-0.1f,-0.3f,-0.5f,-0.7f,-0.9f};
__device__ __constant__ float d_ITS[NK] = {500000.0f,50.f,50.f,50.f,50.f,50.f,50.f,50.f,50.f,50.f,50.f};

// ---------- 1) repack conv weights: w[C][E][k] (f32) -> W[C][Kpad] (f16), kk = j*E + e ----------
__global__ void k_prep_w(const float* __restrict__ wsrc, _Float16* __restrict__ wdst,
                         int kk_len, int KP, int kconv) {
    int idx = blockIdx.x * blockDim.x + threadIdx.x;
    int total = NC * KP;
    if (idx >= total) return;
    int c = idx / KP, kk = idx % KP;
    float v = 0.f;
    if (kk < kk_len) {
        int j = kk / NE, e = kk % NE;
        v = wsrc[(c * NE + e) * kconv + j];
    }
    wdst[idx] = (_Float16)v;
}

// ---------- 2) im2row gather: X[b][pos][Kpad] f16 (half2 per thread) ----------
__global__ void k_im2row(const int* __restrict__ toks, int arrlen, int n_pos,
                         const float* __restrict__ emb, int kconv, int KP,
                         _Float16* __restrict__ dst) {
    int idx = blockIdx.x * blockDim.x + threadIdx.x;
    int KP2 = KP >> 1;
    int total = NB * n_pos * KP2;
    if (idx >= total) return;
    int b   = idx / (n_pos * KP2);
    int r   = idx % (n_pos * KP2);
    int pos = r / KP2;
    int kk  = (r % KP2) * 2;
    float v0 = 0.f, v1 = 0.f;
    if (kk < kconv * NE) {
        int j = kk / NE, e = kk % NE;
        int tp = pos + j;
        if (tp < NLD) {                          // conv right-pad past seq length -> zero vector
            int tok = (tp < arrlen) ? toks[b * arrlen + tp] : 0;  // pad token 0 still embeds
            const float* row = emb + (size_t)tok * NE + e;
            v0 = row[0];
            v1 = row[1];
            __builtin_prefetch(row + NE, 0, 0);
        }
    }
    h2t o; o.x = (_Float16)v0; o.y = (_Float16)v1;
    *(h2t*)(dst + (size_t)idx * 2) = o;
}

// ---------- 3) conv GEMM + bias + ReLU + L2 norm -> gram f16 [pos][128] ----------
// Template NKC (K chunks of 32) so the K-loop fully unrolls; B fragments are
// preloaded into a register array per chunk so loads clause-group ahead of the
// 8-WMMA burst instead of load->wait0->wmma serialization.
template <int NKC>
__global__ __launch_bounds__(32) void k_conv_t(const _Float16* __restrict__ W,
                                               const _Float16* __restrict__ Xd_g,
                                               const _Float16* __restrict__ Xq_g,
                                               const float* __restrict__ bias,
                                               _Float16* __restrict__ Dg_g,
                                               _Float16* __restrict__ Qg_g) {
    constexpr int KP = NKC * 32;
    int id = blockIdx.x;                  // 32 batch * (16 doc tiles + 2 q tiles)
    int b  = id / 18;
    int t  = id % 18;

    const _Float16* Xrow;
    _Float16* Orow;
    if (t < 16) {
        Xrow = Xd_g + ((size_t)b * NLD + t * 16) * KP;
        Orow = Dg_g + ((size_t)b * NLD + t * 16) * NC;
    } else {
        int qt = t - 16;
        Xrow = Xq_g + ((size_t)b * NLQ + qt * 16) * KP;
        Orow = Qg_g + ((size_t)b * NLQ + qt * 16) * NC;
    }

    int lane = threadIdx.x & 31;
    int h = lane >> 4, m = lane & 15;

    v8f zero = {};
    v8f acc[8];
#pragma unroll
    for (int i = 0; i < 8; i++) acc[i] = zero;

#pragma unroll
    for (int kc = 0; kc < NKC; kc++) {
        FragU A;
        const _Float16* a0 = Xrow + (size_t)m * KP + kc * 32 + 16 * h;
        A.u[0] = *(const uint4*)a0;
        A.u[1] = *(const uint4*)(a0 + 8);
        FragU Bm[8];
#pragma unroll
        for (int nt = 0; nt < 8; nt++) {
            const _Float16* bp = W + (size_t)(nt * 16 + m) * KP + kc * 32 + 16 * h;
            Bm[nt].u[0] = *(const uint4*)bp;
            Bm[nt].u[1] = *(const uint4*)(bp + 8);
        }
#pragma unroll
        for (int nt = 0; nt < 8; nt++)
            acc[nt] = __builtin_amdgcn_wmma_f32_16x16x32_f16(
                false, A.v, false, Bm[nt].v, (short)0, acc[nt], false, false);
    }

    // bias + relu
    float vals[8][8];                      // [ntile][j]
#pragma unroll
    for (int nt = 0; nt < 8; nt++) {
        float bb = bias[nt * 16 + m];
#pragma unroll
        for (int j = 0; j < 8; j++) {
            float v = acc[nt][j] + bb;
            vals[nt][j] = v > 0.f ? v : 0.f;
        }
    }
    // per-row L2 norm: butterfly over the 16 lanes of this half (wave32)
#pragma unroll
    for (int j = 0; j < 8; j++) {
        float ssq = 0.f;
#pragma unroll
        for (int nt = 0; nt < 8; nt++) ssq += vals[nt][j] * vals[nt][j];
        ssq += __shfl_xor(ssq, 1);
        ssq += __shfl_xor(ssq, 2);
        ssq += __shfl_xor(ssq, 4);
        ssq += __shfl_xor(ssq, 8);
        float inv = 1.0f / (sqrtf(ssq) + 1e-13f);
        int row = j + 8 * h;               // C-fragment row for this lane
        _Float16* orow = Orow + (size_t)row * NC + m;
#pragma unroll
        for (int nt = 0; nt < 8; nt++) orow[nt * 16] = (_Float16)(vals[nt][j] * inv);
    }
}

// ---------- 4) similarity (32x256 per pair) + RBF + log-pool ----------
// The Q-gram tile (8 KB) is staged into LDS with the CDNA5 async engine
// (global_load_async_to_lds_b128 / s_wait_asynccnt, ASYNCcnt-tracked), then
// A-fragments are served from LDS (ds_load_b128) with 16x reuse.
// RBF masking is branchless (0/1 float fused into the accumulate) so the inner
// loop is straight-line TRANS/VALU work with no exec-mask ladders.
__global__ __launch_bounds__(256) void k_sim(const _Float16* __restrict__ Qg,
                                             const _Float16* __restrict__ Dg,
                                             const int* __restrict__ q_tokens,
                                             const int* __restrict__ d_tokens,
                                             float* __restrict__ feats) {
    __shared__ alignas(16) _Float16 sQ[NLQ * NC];    // 8 KB
    __shared__ float per_kq[NLQ][NK];

    int bid  = blockIdx.x;
    int b    = bid / 9, pair = bid % 9;
    int g    = pair / 3, hg = pair % 3;
    int tid  = threadIdx.x;

    const _Float16* Q = Qg + (size_t)(g  * NB + b) * NLQ * NC;
    const _Float16* D = Dg + (size_t)(hg * NB + b) * NLD * NC;

    // async-stage Q gram into LDS: 512 x 16B segments, 2 per thread
    for (int i = tid; i < (NLQ * NC) / 8; i += 256) {
        unsigned lds_off = (unsigned)(uintptr_t)(&sQ[i * 8]);
        const _Float16* gp = Q + i * 8;
        asm volatile("global_load_async_to_lds_b128 %0, %1, off"
                     :: "v"(lds_off), "v"(gp) : "memory");
    }
    for (int t = tid; t < NLQ * NK; t += 256) ((float*)per_kq)[t] = 0.f;
    asm volatile("s_wait_asynccnt 0x0" ::: "memory");
    __syncthreads();

    int lane = tid & 31, wave = tid >> 5;
    int h = lane >> 4, m = lane & 15;
    int qtile = wave >> 2, dgrp = wave & 3;   // 4 waves per q-tile, 4 doc tiles each

    // loop-invariant q-pad mask (0/1 float) for this lane's C-fragment rows
    float qpf[8];
#pragma unroll
    for (int j = 0; j < 8; j++)
        qpf[j] = (q_tokens[b * NLQ + qtile * 16 + j + 8 * h] > 0) ? 1.f : 0.f;

    float acc[8][NK];
#pragma unroll
    for (int j = 0; j < 8; j++)
#pragma unroll
        for (int k = 0; k < NK; k++) acc[j][k] = 0.f;

    for (int di = 0; di < 4; di++) {
        int dtile = dgrp + di * 4;
        v8f c = {};
#pragma unroll
        for (int kc = 0; kc < 4; kc++) {
            FragU A, Bm;
            const _Float16* ap = &sQ[(qtile * 16 + m) * NC + kc * 32 + 16 * h];
            A.u[0] = *(const uint4*)ap;
            A.u[1] = *(const uint4*)(ap + 8);
            const _Float16* bp = D + (size_t)(dtile * 16 + m) * NC + kc * 32 + 16 * h;
            Bm.u[0] = *(const uint4*)bp;
            Bm.u[1] = *(const uint4*)(bp + 8);
            c = __builtin_amdgcn_wmma_f32_16x16x32_f16(
                false, A.v, false, Bm.v, (short)0, c, false, false);
        }
        int dcol = dtile * 16 + m;
        float dpf = (d_tokens[b * NLD + dcol] > 0) ? 1.f : 0.f;
        // branchless masked RBF accumulation; |c[j]| <= 1 so __expf is safe
        // even for masked-out positions.
#pragma unroll
        for (int j = 0; j < 8; j++) {
            float w  = dpf * qpf[j];
            float mv = c[j];
#pragma unroll
            for (int k = 0; k < NK; k++) {
                float d0 = mv - d_MU[k];
                acc[j][k] = fmaf(w, __expf(-d0 * d0 * d_ITS[k]), acc[j][k]);
            }
        }
    }
    // sum over the 16 doc columns held by this half, then merge across waves in LDS
#pragma unroll
    for (int j = 0; j < 8; j++) {
        int qrow = qtile * 16 + j + 8 * h;
#pragma unroll
        for (int k = 0; k < NK; k++) {
            float v = acc[j][k];
            v += __shfl_xor(v, 1);
            v += __shfl_xor(v, 2);
            v += __shfl_xor(v, 4);
            v += __shfl_xor(v, 8);
            if (m == 0) atomicAdd(&per_kq[qrow][k], v);
        }
    }
    __syncthreads();

    if (tid < NLQ * NK) {
        int q = tid / NK, k = tid % NK;
        float v = per_kq[q][k];
        v = logf(fmaxf(v, 1e-10f)) * 0.01f;
        float oov = (q_tokens[b * NLQ + q] > 1) ? 1.f : 0.f;
        per_kq[q][k] = v * oov;
    }
    __syncthreads();

    if (tid < NK) {
        float s = 0.f;
        for (int q = 0; q < NLQ; q++) s += per_kq[q][tid];
        feats[(size_t)b * 99 + pair * NK + tid] = s;
    }
}

// ---------- 5) dense + tanh ----------
__global__ void k_final(const float* __restrict__ feats, const float* __restrict__ dw,
                        float* __restrict__ out) {
    int b = threadIdx.x;
    if (b >= NB) return;
    float s = 0.f;
    for (int i = 0; i < 99; i++) s += feats[b * 99 + i] * dw[i];
    out[b] = tanhf(s);
}

extern "C" void kernel_launch(void* const* d_in, const int* in_sizes, int n_in,
                              void* d_out, int out_size, void* d_ws, size_t ws_size,
                              hipStream_t stream) {
    (void)in_sizes; (void)n_in; (void)out_size; (void)ws_size;
    const int*   q_tokens = (const int*)d_in[0];
    const int*   d_tokens = (const int*)d_in[1];
    const float* emb      = (const float*)d_in[2];
    const float* w1       = (const float*)d_in[3];
    const float* cb1      = (const float*)d_in[4];
    const float* w2       = (const float*)d_in[5];
    const float* cb2      = (const float*)d_in[6];
    const float* w3       = (const float*)d_in[7];
    const float* cb3      = (const float*)d_in[8];
    const float* dw       = (const float*)d_in[9];
    float* out = (float*)d_out;

    char* ws = (char*)d_ws;
    float*     feats = (float*)(ws + OFF_FEATS);
    _Float16*  Wall  = (_Float16*)(ws + OFF_W);
    _Float16*  Xd    = (_Float16*)(ws + OFF_XD);
    _Float16*  Xq    = (_Float16*)(ws + OFF_XQ);
    _Float16*  Dg    = (_Float16*)(ws + OFF_DG);
    _Float16*  Qg    = (_Float16*)(ws + OFF_QG);

    // 1) weight repack (f32 -> f16 fragment-friendly [C][Kpad])
    k_prep_w<<<(NC * 320 + 255) / 256, 256, 0, stream>>>(w1, Wall + 0,      300, 320, 1);
    k_prep_w<<<(NC * 608 + 255) / 256, 256, 0, stream>>>(w2, Wall + 40960,  600, 608, 2);
    k_prep_w<<<(NC * 928 + 255) / 256, 256, 0, stream>>>(w3, Wall + 118784, 900, 928, 3);

    // 2) im2row gathers (doc: 256 pos, arrlen 256; query: 32 pos, arrlen 32)
    k_im2row<<<(NB * NLD * 160 + 255) / 256, 256, 0, stream>>>(d_tokens, NLD, NLD, emb, 1, 320, Xd + 0);
    k_im2row<<<(NB * NLD * 304 + 255) / 256, 256, 0, stream>>>(d_tokens, NLD, NLD, emb, 2, 608, Xd + 2621440);
    k_im2row<<<(NB * NLD * 464 + 255) / 256, 256, 0, stream>>>(d_tokens, NLD, NLD, emb, 3, 928, Xd + 7602176);
    k_im2row<<<(NB * NLQ * 160 + 255) / 256, 256, 0, stream>>>(q_tokens, NLQ, NLQ, emb, 1, 320, Xq + 0);
    k_im2row<<<(NB * NLQ * 304 + 255) / 256, 256, 0, stream>>>(q_tokens, NLQ, NLQ, emb, 2, 608, Xq + 327680);
    k_im2row<<<(NB * NLQ * 464 + 255) / 256, 256, 0, stream>>>(q_tokens, NLQ, NLQ, emb, 3, 928, Xq + 950272);

    // 3) conv GEMM + ReLU + L2 norm, one specialized launch per gram (K fully unrolled)
    k_conv_t<10><<<NB * 18, 32, 0, stream>>>(Wall + 0,      Xd + 0,       Xq + 0,      cb1,
                                             Dg + 0,                      Qg + 0);
    k_conv_t<19><<<NB * 18, 32, 0, stream>>>(Wall + 40960,  Xd + 2621440, Xq + 327680, cb2,
                                             Dg + 1 * NB * NLD * NC,      Qg + 1 * NB * NLQ * NC);
    k_conv_t<29><<<NB * 18, 32, 0, stream>>>(Wall + 118784, Xd + 7602176, Xq + 950272, cb3,
                                             Dg + 2 * NB * NLD * NC,      Qg + 2 * NB * NLQ * NC);

    // 4) similarity + RBF pooling (32 batch * 9 gram pairs)
    k_sim<<<NB * 9, 256, 0, stream>>>(Qg, Dg, q_tokens, d_tokens, feats);

    // 5) dense + tanh
    k_final<<<1, 32, 0, stream>>>(feats, dw, out);
}
